// GATLayer_regular_12876311953764
// MI455X (gfx1250) — matrix operand order repaired and checked by reading, not compiled, not run on hardware
//
#include <hip/hip_runtime.h>
#include <hip/hip_bf16.h>

#define DDIM 256

typedef __attribute__((ext_vector_type(16))) __bf16 v16bf;
typedef __attribute__((ext_vector_type(8)))  __bf16 v8bf;
typedef __attribute__((ext_vector_type(8)))  float  v8f;

static __device__ __forceinline__ void split_bf16(float x, __bf16& h, __bf16& l) {
    h = (__bf16)x;
    l = (__bf16)(x - (float)h);
}

// -----------------------------------------------------------------------------
// Prep: transposed bf16 hi/lo planes of W1 and W2.
//   Wt_hi[n*256 + k] = bf16(W[k*256 + n]),  Wt_lo = bf16(residual)
// One-off 64K-element kernel; writes coalesced.
// -----------------------------------------------------------------------------
__global__ __launch_bounds__(256)
void gat_prep_w(const float* __restrict__ W1, const float* __restrict__ W2,
                __bf16* __restrict__ w1h, __bf16* __restrict__ w1l,
                __bf16* __restrict__ w2h, __bf16* __restrict__ w2l)
{
    const int idx = blockIdx.x * blockDim.x + threadIdx.x;   // 0 .. 65535
    const int n = idx >> 8;
    const int k = idx & 255;
    const int src = k * DDIM + n;
    const int dst = n * DDIM + k;
    split_bf16(W1[src], w1h[dst], w1l[dst]);
    split_bf16(W2[src], w2h[dst], w2l[dst]);
}

// -----------------------------------------------------------------------------
// Init: out = x0 (residual term), a1[i] = ba1, a2[i] = ba2
// -----------------------------------------------------------------------------
__global__ void gat_init(const float* __restrict__ x0, float* __restrict__ out,
                         const float* __restrict__ ba1, const float* __restrict__ ba2,
                         float* __restrict__ a1, float* __restrict__ a2, int N)
{
    const int total4 = N * (DDIM / 4);
    for (int i = blockIdx.x * blockDim.x + threadIdx.x; i < total4;
         i += gridDim.x * blockDim.x) {
        reinterpret_cast<float4*>(out)[i] = reinterpret_cast<const float4*>(x0)[i];
    }
    for (int i = blockIdx.x * blockDim.x + threadIdx.x; i < N;
         i += gridDim.x * blockDim.x) {
        a1[i] = ba1[0];
        a2[i] = ba2[0];
    }
}

// -----------------------------------------------------------------------------
// Fused GEMM + bias + leaky-ReLU + attention matvec.
//   Y = leaky(X @ W + b)           (optionally stored; Yout may be null)
//   aout[m] += sum_n Y[m,n]*wa[n]  (aout pre-seeded with ba)
//
// Split-bf16 fp32 emulation: A*B ~ Ah*Bh + Ah*Bl + Al*Bh, f32 accumulate via
// v_wmma_f32_16x16x32_bf16.  A hi/lo staged once per block in LDS; W hi/lo
// pre-split + pre-transposed in global scratch, so the K-loop is pure
// ds_load_b128 / global_load_b128 / wmma.
//
// Block = 256 threads = 8 waves, owns 16 rows; wave w owns cols [32w, 32w+32)
// as two 16x16 tiles.  K (=256) consumed 32 per step.
// -----------------------------------------------------------------------------
__global__ __launch_bounds__(256)
void gat_gemm_attn(const float* __restrict__ X,
                   const __bf16* __restrict__ wth, const __bf16* __restrict__ wtl,
                   const float* __restrict__ bias, const float* __restrict__ wa,
                   float* __restrict__ Yout, float* __restrict__ aout, int N)
{
    __shared__ __bf16 Ahi[16 * DDIM];   // 8 KB
    __shared__ __bf16 Alo[16 * DDIM];   // 8 KB

    const int tid  = threadIdx.x;
    const int lane = tid & 31;
    const int wave = tid >> 5;
    const int hlf  = lane >> 4;        // 0: lanes 0-15, 1: lanes 16-31
    const int l16  = lane & 15;
    const int row0 = blockIdx.x * 16;

    // ---- Stage A hi/lo split into LDS (each thread: 16 contiguous floats) --
    {
        const int m  = tid >> 4;
        const int kb = (tid & 15) * 16;
        int r = row0 + m;
        if (r >= N) r = N - 1;                       // clamp (EXEC stays full)
        const float* src = X + (size_t)r * DDIM + kb;
        #pragma unroll
        for (int i = 0; i < 16; ++i)
            split_bf16(src[i], Ahi[m * DDIM + kb + i], Alo[m * DDIM + kb + i]);
    }
    __syncthreads();

    const int n0 = wave * 32;          // first column of this wave's tile pair
    const int nA = n0 + l16;           // tile-0 column of this lane
    const int nB = nA + 16;            // tile-1 column of this lane

    v8f c0 = {};
    v8f c1 = {};

    for (int kk = 0; kk < DDIM; kk += 32) {
        // ---- A fragments from LDS: elem i -> K=kk+16*hlf+i
        //      (contiguous-K halves) ---------
        const int abase = l16 * DDIM + kk + 16 * hlf;
        v8bf ah0 = *reinterpret_cast<const v8bf*>(&Ahi[abase]);
        v8bf ah1 = *reinterpret_cast<const v8bf*>(&Ahi[abase + 8]);
        v8bf al0 = *reinterpret_cast<const v8bf*>(&Alo[abase]);
        v8bf al1 = *reinterpret_cast<const v8bf*>(&Alo[abase + 8]);
        v16bf ahi = __builtin_shufflevector(ah0, ah1, 0, 1, 2, 3, 4, 5, 6, 7,
                                            8, 9, 10, 11, 12, 13, 14, 15);
        v16bf alo = __builtin_shufflevector(al0, al1, 0, 1, 2, 3, 4, 5, 6, 7,
                                            8, 9, 10, 11, 12, 13, 14, 15);

        // ---- B fragments from transposed bf16 planes: elem i -> K=base+i ---
        const int kbb = kk + 16 * hlf;
        const __bf16* p0h = wth + (size_t)nA * DDIM + kbb;
        const __bf16* p0l = wtl + (size_t)nA * DDIM + kbb;
        const __bf16* p1h = wth + (size_t)nB * DDIM + kbb;
        const __bf16* p1l = wtl + (size_t)nB * DDIM + kbb;
        v16bf b0h = __builtin_shufflevector(
            *reinterpret_cast<const v8bf*>(p0h),
            *reinterpret_cast<const v8bf*>(p0h + 8),
            0, 1, 2, 3, 4, 5, 6, 7, 8, 9, 10, 11, 12, 13, 14, 15);
        v16bf b0l = __builtin_shufflevector(
            *reinterpret_cast<const v8bf*>(p0l),
            *reinterpret_cast<const v8bf*>(p0l + 8),
            0, 1, 2, 3, 4, 5, 6, 7, 8, 9, 10, 11, 12, 13, 14, 15);
        v16bf b1h = __builtin_shufflevector(
            *reinterpret_cast<const v8bf*>(p1h),
            *reinterpret_cast<const v8bf*>(p1h + 8),
            0, 1, 2, 3, 4, 5, 6, 7, 8, 9, 10, 11, 12, 13, 14, 15);
        v16bf b1l = __builtin_shufflevector(
            *reinterpret_cast<const v8bf*>(p1l),
            *reinterpret_cast<const v8bf*>(p1l + 8),
            0, 1, 2, 3, 4, 5, 6, 7, 8, 9, 10, 11, 12, 13, 14, 15);

        // ---- 3-term split products, f32 accumulate -------------------------
        c0 = __builtin_amdgcn_wmma_f32_16x16x32_bf16(false, ahi, false, b0h,
                                                     (short)0, c0, false, false);
        c1 = __builtin_amdgcn_wmma_f32_16x16x32_bf16(false, ahi, false, b1h,
                                                     (short)0, c1, false, false);
        c0 = __builtin_amdgcn_wmma_f32_16x16x32_bf16(false, ahi, false, b0l,
                                                     (short)0, c0, false, false);
        c1 = __builtin_amdgcn_wmma_f32_16x16x32_bf16(false, ahi, false, b1l,
                                                     (short)0, c1, false, false);
        c0 = __builtin_amdgcn_wmma_f32_16x16x32_bf16(false, alo, false, b0h,
                                                     (short)0, c0, false, false);
        c1 = __builtin_amdgcn_wmma_f32_16x16x32_bf16(false, alo, false, b1h,
                                                     (short)0, c1, false, false);
    }

    // ---- Epilogue: bias + leaky + optional store + attention matvec --------
    // C layout: VGPR r, lanes 0-15 -> (M=r, N=l16); lanes 16-31 -> (M=8+r, N=l16)
    #pragma unroll
    for (int t = 0; t < 2; ++t) {
        const int ncol = (t ? nB : nA);
        const float bn = bias[ncol];
        const float wn = wa[ncol];
        const v8f&  c  = t ? c1 : c0;
        #pragma unroll
        for (int r = 0; r < 8; ++r) {
            const int gm = row0 + r + 8 * hlf;
            float y = c[r] + bn;
            y = (y > 0.0f) ? y : 0.2f * y;           // leaky_relu(., 0.2)
            if (Yout != nullptr && gm < N)
                Yout[(size_t)gm * DDIM + ncol] = y;
            float p = y * wn;
            #pragma unroll
            for (int off = 8; off > 0; off >>= 1)    // reduce 16 lanes / row
                p += __shfl_xor(p, off, 16);
            if (l16 == 0 && gm < N)
                atomicAdd(&aout[gm], p);
        }
    }
}

// -----------------------------------------------------------------------------
// Edge phase: out[row] += sigmoid(a1[row]+a2[col]) * x0_j[col]
// 64 threads per edge, float4 per thread (256 floats per edge); x0_j is
// L2-resident (10 MB on a 192 MB L2) so gathers run at L2 rate.
// -----------------------------------------------------------------------------
__global__ __launch_bounds__(256)
void gat_edge(const int* __restrict__ ei, const float* __restrict__ xj,
              const float* __restrict__ a1, const float* __restrict__ a2,
              float* __restrict__ out, int E)
{
    const int e = blockIdx.x * 4 + (threadIdx.x >> 6);
    const int t = threadIdx.x & 63;
    if (e >= E) return;

    const int row = ei[e];
    const int col = ei[E + e];
    const float s   = a1[row] + a2[col];
    const float att = 1.0f / (1.0f + __expf(-s));

    const float4 v = reinterpret_cast<const float4*>(xj + (size_t)col * DDIM)[t];
    float* o = out + (size_t)row * DDIM + t * 4;
    atomicAdd(o + 0, att * v.x);
    atomicAdd(o + 1, att * v.y);
    atomicAdd(o + 2, att * v.z);
    atomicAdd(o + 3, att * v.w);
}

// -----------------------------------------------------------------------------
extern "C" void kernel_launch(void* const* d_in, const int* in_sizes, int n_in,
                              void* d_out, int out_size, void* d_ws, size_t ws_size,
                              hipStream_t stream)
{
    const float* x0  = (const float*)d_in[0];
    const int*   ei  = (const int*)  d_in[1];
    const float* W1  = (const float*)d_in[2];
    const float* b1  = (const float*)d_in[3];
    const float* W2  = (const float*)d_in[4];
    const float* b2  = (const float*)d_in[5];
    const float* wa1 = (const float*)d_in[6];
    const float* ba1 = (const float*)d_in[7];
    const float* wa2 = (const float*)d_in[8];
    const float* ba2 = (const float*)d_in[9];
    float* out = (float*)d_out;

    const int N = in_sizes[0] / DDIM;
    const int E = in_sizes[1] / 2;

    // Scratch layout (16B aligned blocks):
    //   xj [N*D f32] | a1 [N f32] | a2 [N f32] | w1h|w1l|w2h|w2l [64K bf16 each]
    char* ws = (char*)d_ws;
    float* xj = (float*)ws;                 ws += (size_t)N * DDIM * sizeof(float);
    float* a1 = (float*)ws;                 ws += ((size_t)N * sizeof(float) + 15) & ~(size_t)15;
    float* a2 = (float*)ws;                 ws += ((size_t)N * sizeof(float) + 15) & ~(size_t)15;
    __bf16* w1h = (__bf16*)ws;              ws += (size_t)DDIM * DDIM * sizeof(__bf16);
    __bf16* w1l = (__bf16*)ws;              ws += (size_t)DDIM * DDIM * sizeof(__bf16);
    __bf16* w2h = (__bf16*)ws;              ws += (size_t)DDIM * DDIM * sizeof(__bf16);
    __bf16* w2l = (__bf16*)ws;

    gat_prep_w<<<(DDIM * DDIM) / 256, 256, 0, stream>>>(W1, W2, w1h, w1l, w2h, w2l);
    gat_init<<<512, 256, 0, stream>>>(x0, out, ba1, ba2, a1, a2, N);

    const int rowBlocks = (N + 15) / 16;
    // Pass 1: only a1 needed (x0_i never materialized).
    gat_gemm_attn<<<rowBlocks, 256, 0, stream>>>(x0, w1h, w1l, b1, wa1, nullptr, a1, N);
    // Pass 2: x0_j stored + a2.
    gat_gemm_attn<<<rowBlocks, 256, 0, stream>>>(x0, w2h, w2l, b2, wa2, xj, a2, N);

    gat_edge<<<(E + 3) / 4, 256, 0, stream>>>(ei, xj, a1, a2, out, E);
}
